// DecoderLayer_2044404433630
// MI455X (gfx1250) — matrix operand order repaired and checked by reading, not compiled, not run on hardware
//
#include <hip/hip_runtime.h>

// ---------------------------------------------------------------------------
// Transformer decoder layer for MI455X (gfx1250, wave32, WMMA).
// All GEMMs use v_wmma_f32_16x16x32_bf16 (bf16 in, fp32 accumulate).
// - softmax scale * log2(e) folded into Q projection epilogue (exp2 softmax)
// - V projection writes per-head TRANSPOSED output Vt[(b,h,d), seq] so the
//   flash PV B-fragments are contiguous 32B loads (no scalar gather)
// - flash waves process 32 query rows: halves K/V streaming traffic,
//   18 WMMAs per 16 b128 loads per 32-key step
// - P row-sums ride in extra WMMA accumulators (P x ones)
// - mask loads skipped when a precomputed flag says the mask is all-ones
// ---------------------------------------------------------------------------

#define D_MODEL   1024
#define NUM_HEADS 16
#define D_FF      4096
#define SEQ       2048
#define BATCH     4
#define MROWS     (BATCH * SEQ)   // 8192

// scores use exp2: fold 1/sqrt(64) * log2(e) into Q projection epilogue
#define QSCALE    (0.125f * 1.4426950408889634f)

typedef __attribute__((ext_vector_type(16))) __bf16 v16bf;
typedef __attribute__((ext_vector_type(8)))  __bf16 v8bf;
typedef __attribute__((ext_vector_type(8)))  float  v8f;

__device__ __forceinline__ __bf16 f2bf(float f) {
  union { float f; unsigned u; } a; a.f = f;
  unsigned r = (a.u + 0x7FFFu + ((a.u >> 16) & 1u)) >> 16;   // round-nearest-even
  union { unsigned short s; __bf16 b; } o; o.s = (unsigned short)r;
  return o.b;
}

__device__ __forceinline__ v8f zero8() {
  v8f z = {0.f, 0.f, 0.f, 0.f, 0.f, 0.f, 0.f, 0.f};
  return z;
}

__device__ __forceinline__ v16bf cat8(v8bf lo, v8bf hi) {
  return __builtin_shufflevector(lo, hi, 0,1,2,3,4,5,6,7,8,9,10,11,12,13,14,15);
}

// A fragment: 16(M) x 32(K) bf16 from row-major [., ld].
__device__ __forceinline__ v16bf load_a_frag(const __bf16* A, int ld, size_t mBase,
                                             int kBase, int r, int h) {
  const __bf16* p = A + (mBase + (size_t)r) * ld + kBase + h * 16;
  v8bf lo = *(const v8bf*)p;
  v8bf hi = *(const v8bf*)(p + 8);
  return cat8(lo, hi);
}

// B fragment (NT): 32(K) x 16(N) from row-major [N, K] (k-contiguous rows).
__device__ __forceinline__ v16bf load_b_frag_nt(const __bf16* W, int ld, size_t nBase,
                                                int kBase, int r, int h) {
  const __bf16* p = W + (nBase + (size_t)r) * ld + kBase + h * 16;
  return *(const v16bf*)p;
}

__device__ __forceinline__ v8f wmma_bf16(v16bf a, v16bf b, v8f c) {
  return __builtin_amdgcn_wmma_f32_16x16x32_bf16(false, a, false, b, (short)0, c,
                                                 false, false);
}

// ---------------------------------------------------------------------------
// fp32 -> bf16 conversion, 4 elements / thread
// ---------------------------------------------------------------------------
__global__ void __launch_bounds__(256) cvt_f32_bf16_kernel(const float* __restrict__ in,
                                                           __bf16* __restrict__ out,
                                                           size_t n4) {
  size_t i = (size_t)blockIdx.x * 256 + threadIdx.x;
  size_t stride = (size_t)gridDim.x * 256;
  for (; i < n4; i += stride) {
    float4 f = ((const float4*)in)[i];
    union { __bf16 b[4]; uint2 u; } o;
    o.b[0] = f2bf(f.x); o.b[1] = f2bf(f.y); o.b[2] = f2bf(f.z); o.b[3] = f2bf(f.w);
    ((uint2*)out)[i] = o.u;
  }
}

// ---------------------------------------------------------------------------
// Mask pre-scan: flag stays 1 iff every mask element is nonzero.
// ---------------------------------------------------------------------------
__global__ void flag_set_kernel(int* f) {
  if (threadIdx.x == 0 && blockIdx.x == 0) { f[0] = 1; f[1] = 1; }
}

__global__ void __launch_bounds__(256) mask_check_kernel(const int* __restrict__ mask,
                                                         size_t n, int* __restrict__ flag) {
  size_t i = (size_t)blockIdx.x * 256 + threadIdx.x;
  size_t stride = (size_t)gridDim.x * 256;
  int ok = 1;
  for (; i < n; i += stride)
    if (mask[i] == 0) ok = 0;
  if (!ok) flag[0] = 0;   // benign race: all writers store 0
}

// ---------------------------------------------------------------------------
// C[M,N] = (A[M,K] * W[N,K]^T + bias) * scale   (bf16 in, fp32 accumulate)
// Block: 128 threads (4 waves), tile 128x128. Wave tile 64x64 (4x4 frags).
// OUTMODE: 0 = f32, 1 = bf16, 2 = bf16 per-head transposed Vt[(b,h,d), seq]
//          (mode 2 assumes M = MROWS, N = D_MODEL)
// ---------------------------------------------------------------------------
template <int RELU, int OUTMODE>
__global__ void __launch_bounds__(128) gemm_nt_kernel(
    const __bf16* __restrict__ A, const __bf16* __restrict__ W,
    const float* __restrict__ bias, float* __restrict__ Cf,
    __bf16* __restrict__ Cb, int M, int N, int K, float scale) {
  const int lane = threadIdx.x & 31;
  const int wave = threadIdx.x >> 5;
  const int r = lane & 15, h = lane >> 4;
  const size_t mBase = (size_t)blockIdx.y * 128 + (size_t)(wave >> 1) * 64;
  const size_t nBase = (size_t)blockIdx.x * 128 + (size_t)(wave & 1) * 64;

  v8f acc[4][4];
#pragma unroll
  for (int i = 0; i < 4; i++)
#pragma unroll
    for (int j = 0; j < 4; j++) acc[i][j] = zero8();

#pragma unroll 2
  for (int k = 0; k < K; k += 32) {
    if (k + 64 < K) {  // prefetch ahead (global_prefetch_b8)
      __builtin_prefetch(A + (mBase + r) * (size_t)K + k + 64, 0, 1);
      __builtin_prefetch(W + (nBase + r) * (size_t)K + k + 64, 0, 1);
    }
    v16bf a[4], b[4];
#pragma unroll
    for (int i = 0; i < 4; i++) a[i] = load_a_frag(A, K, mBase + 16 * i, k, r, h);
#pragma unroll
    for (int j = 0; j < 4; j++) b[j] = load_b_frag_nt(W, K, nBase + 16 * j, k, r, h);
#pragma unroll
    for (int i = 0; i < 4; i++)
#pragma unroll
      for (int j = 0; j < 4; j++) acc[i][j] = wmma_bf16(a[i], b[j], acc[i][j]);
  }

  float bv[4];
#pragma unroll
  for (int j = 0; j < 4; j++) bv[j] = bias ? bias[nBase + j * 16 + r] : 0.f;

  if (OUTMODE == 2) {
    // transposed V output: Vt[((b*16 + head)*64 + dl) * SEQ + s]
    // for fixed (i,j): the 8 accumulator rows are consecutive s -> one 16B store
#pragma unroll
    for (int i = 0; i < 4; i++) {
      size_t sTile = mBase + (size_t)i * 16 + 8 * h;  // multiple of 8
      size_t bIdx = sTile >> 11;                       // batch index (SEQ = 2048)
      size_t s = sTile & 2047;
#pragma unroll
      for (int j = 0; j < 4; j++) {
        size_t colg = nBase + (size_t)j * 16 + r;      // 0..1023
        size_t headi = colg >> 6, dl = colg & 63;
        union { __bf16 b[8]; v8bf v8; } pk;
#pragma unroll
        for (int v = 0; v < 8; v++)
          pk.b[v] = f2bf((acc[i][j][v] + bv[j]) * scale);
        *(v8bf*)(Cb + ((bIdx * 16 + headi) * 64 + dl) * (size_t)SEQ + s) = pk.v8;
      }
    }
  } else {
#pragma unroll
    for (int i = 0; i < 4; i++) {
#pragma unroll
      for (int j = 0; j < 4; j++) {
#pragma unroll
        for (int v = 0; v < 8; v++) {
          size_t row = mBase + (size_t)i * 16 + v + 8 * h;  // C layout: row = v + 8h
          size_t col = nBase + (size_t)j * 16 + r;          //           col = r
          float val = (acc[i][j][v] + bv[j]) * scale;
          if (RELU) val = fmaxf(val, 0.f);
          if (OUTMODE == 1) Cb[row * (size_t)N + col] = f2bf(val);
          else              Cf[row * (size_t)N + col] = val;
        }
      }
    }
  }
}

// ---------------------------------------------------------------------------
// Flash attention: one wave per 32-query block per (batch, head).
// Q pre-scaled by 0.125*log2(e); softmax uses native base-2 exp.
// Vt is per-head transposed: Vt[((b*16+head)*64 + d) * Sk + s]  -> contiguous
// 32B B-fragment loads shared by both 16-row query sub-tiles.
// ---------------------------------------------------------------------------
__global__ void __launch_bounds__(32) flash_attn_kernel(
    const __bf16* __restrict__ Q, const __bf16* __restrict__ Kmat,
    const __bf16* __restrict__ Vt, const int* __restrict__ mask,
    int maskPerQuery, const int* __restrict__ maskFlag,
    __bf16* __restrict__ O, int Sq, int Sk) {
  __shared__ __attribute__((aligned(16))) __bf16 Plds[2][16 * 32];

  const int lane = threadIdx.x;
  const int r = lane & 15, h = lane >> 4;
  const int bh = blockIdx.y;
  const int b = bh >> 4, head = bh & 15;
  const int qb = blockIdx.x;                 // 32-row query block
  const bool noMask = (maskFlag[0] != 0);    // uniform branch

  const __bf16* Qh = Q + head * 64;
  const __bf16* Kh = Kmat + head * 64;
  const __bf16* Vbh = Vt + ((size_t)(b * 16 + head) * 64) * (size_t)Sk;
  const size_t qRow = (size_t)b * Sq + (size_t)qb * 32;

  v16bf qa[2][2];
#pragma unroll
  for (int s2 = 0; s2 < 2; s2++) {
    qa[s2][0] = load_a_frag(Qh, D_MODEL, qRow + s2 * 16, 0, r, h);
    qa[s2][1] = load_a_frag(Qh, D_MODEL, qRow + s2 * 16, 32, r, h);
  }

  v16bf ones;
  {
    __bf16 one = f2bf(1.0f);
#pragma unroll
    for (int e = 0; e < 16; e++) ones[e] = one;
  }

  float m[2][8];
  v8f o[2][4], osum[2];
#pragma unroll
  for (int s2 = 0; s2 < 2; s2++) {
    osum[s2] = zero8();
#pragma unroll
    for (int v = 0; v < 8; v++) m[s2][v] = -1e30f;
#pragma unroll
    for (int c4 = 0; c4 < 4; c4++) o[s2][c4] = zero8();
  }

#pragma unroll 1
  for (int j = 0; j < Sk; j += 32) {
    const size_t kRow = (size_t)b * Sk + j;
    if (j + 32 < Sk) {
      __builtin_prefetch(Kh + (kRow + 32 + r) * (size_t)D_MODEL, 0, 1);
      __builtin_prefetch(Vbh + (size_t)(h * 32 + r * 2) * Sk + j + 32, 0, 1);
    }
    // K fragments shared by both query sub-tiles
    v16bf kb00 = load_b_frag_nt(Kh, D_MODEL, kRow +  0,  0, r, h);
    v16bf kb01 = load_b_frag_nt(Kh, D_MODEL, kRow +  0, 32, r, h);
    v16bf kb10 = load_b_frag_nt(Kh, D_MODEL, kRow + 16,  0, r, h);
    v16bf kb11 = load_b_frag_nt(Kh, D_MODEL, kRow + 16, 32, r, h);

#pragma unroll
    for (int s2 = 0; s2 < 2; s2++) {
      v8f s0 = zero8(), s1 = zero8();
      s0 = wmma_bf16(qa[s2][0], kb00, s0); s0 = wmma_bf16(qa[s2][1], kb01, s0);
      s1 = wmma_bf16(qa[s2][0], kb10, s1); s1 = wmma_bf16(qa[s2][1], kb11, s1);

      float corr[8];
#pragma unroll
      for (int v = 0; v < 8; v++) {
        float a0 = s0[v];   // already scaled (QSCALE folded into Q projection)
        float a1 = s1[v];
        if (!noMask) {
          const int qi = qb * 32 + s2 * 16 + v + 8 * h;
          const int* mrow = maskPerQuery ? (mask + ((size_t)b * Sq + qi) * Sk)
                                         : (mask + (size_t)b * Sk);
          if (mrow[j + r] == 0)      a0 = -1e9f;
          if (mrow[j + 16 + r] == 0) a1 = -1e9f;
        }
        float mx = fmaxf(a0, a1);
#pragma unroll
        for (int off = 1; off < 16; off <<= 1) mx = fmaxf(mx, __shfl_xor(mx, off, 32));
        float mn = fmaxf(m[s2][v], mx);
        corr[v] = __builtin_amdgcn_exp2f(m[s2][v] - mn);
        m[s2][v] = mn;
        // P -> LDS (bf16), C layout row = v + 8h, col = r / r+16
        Plds[s2][(v + 8 * h) * 32 + r]      = f2bf(__builtin_amdgcn_exp2f(a0 - mn));
        Plds[s2][(v + 8 * h) * 32 + 16 + r] = f2bf(__builtin_amdgcn_exp2f(a1 - mn));
      }
      // rescale O and running-sum accumulators (same row layout as stats)
#pragma unroll
      for (int v = 0; v < 8; v++) {
        osum[s2][v] *= corr[v];
#pragma unroll
        for (int c4 = 0; c4 < 4; c4++) o[s2][c4][v] *= corr[v];
      }
    }

    asm volatile("s_wait_dscnt 0" ::: "memory");  // LDS in-order per wave; keep compiler honest
    v16bf pa0, pa1;
    {
      v8bf lo0 = *(const v8bf*)(&Plds[0][0] + r * 32 + h * 16);
      v8bf hi0 = *(const v8bf*)(&Plds[0][0] + r * 32 + 8 + h * 16);
      pa0 = cat8(lo0, hi0);
      v8bf lo1 = *(const v8bf*)(&Plds[1][0] + r * 32 + h * 16);
      v8bf hi1 = *(const v8bf*)(&Plds[1][0] + r * 32 + 8 + h * 16);
      pa1 = cat8(lo1, hi1);
    }

    // O[32,64] += P[32,32] x V[32,64]; V B-frags contiguous from Vt, shared
#pragma unroll
    for (int c4 = 0; c4 < 4; c4++) {
      v16bf vb = *(const v16bf*)(Vbh + (size_t)(c4 * 16 + r) * Sk + j + h * 16);
      o[0][c4] = wmma_bf16(pa0, vb, o[0][c4]);
      o[1][c4] = wmma_bf16(pa1, vb, o[1][c4]);
    }
    osum[0] = wmma_bf16(pa0, ones, osum[0]);
    osum[1] = wmma_bf16(pa1, ones, osum[1]);
  }

  // normalize and write O (bf16). osum holds row sums (all cols equal).
#pragma unroll
  for (int s2 = 0; s2 < 2; s2++) {
#pragma unroll
    for (int v = 0; v < 8; v++) {
      float l = osum[s2][v];
      float inv = (l > 0.f) ? 1.0f / l : 0.f;
      size_t row = (size_t)b * Sq + qb * 32 + s2 * 16 + v + 8 * h;
#pragma unroll
      for (int c4 = 0; c4 < 4; c4++)
        O[row * D_MODEL + head * 64 + c4 * 16 + r] = f2bf(o[s2][c4][v] * inv);
    }
  }
}

// ---------------------------------------------------------------------------
// y = LayerNorm(x + t) * g + b ; writes f32 (residual chain) and optional bf16
// One row (1024 elems) per 256-thread block; float4 vectorized.
// ---------------------------------------------------------------------------
__global__ void __launch_bounds__(256) residual_ln_kernel(
    const float* __restrict__ x, const float* __restrict__ t,
    const float* __restrict__ g, const float* __restrict__ bta,
    float* __restrict__ yf, __bf16* __restrict__ yb) {
  __shared__ float red[256];
  const int row = blockIdx.x;
  const int tid = threadIdx.x;
  const size_t base = (size_t)row * D_MODEL;

  float4 xv = ((const float4*)(x + base))[tid];
  float4 tv = ((const float4*)(t + base))[tid];
  float v[4] = {xv.x + tv.x, xv.y + tv.y, xv.z + tv.z, xv.w + tv.w};

  red[tid] = v[0] + v[1] + v[2] + v[3];
  __syncthreads();
  for (int o = 128; o > 0; o >>= 1) {
    if (tid < o) red[tid] += red[tid + o];
    __syncthreads();
  }
  float mu = red[0] * (1.0f / D_MODEL);
  __syncthreads();

  float sq = 0.f;
#pragma unroll
  for (int i = 0; i < 4; i++) { float d = v[i] - mu; sq += d * d; }
  red[tid] = sq;
  __syncthreads();
  for (int o = 128; o > 0; o >>= 1) {
    if (tid < o) red[tid] += red[tid + o];
    __syncthreads();
  }
  float rstd = rsqrtf(red[0] * (1.0f / D_MODEL) + 1e-5f);

  float4 gv = ((const float4*)g)[tid];
  float4 bv = ((const float4*)bta)[tid];
  float4 y;
  y.x = (v[0] - mu) * rstd * gv.x + bv.x;
  y.y = (v[1] - mu) * rstd * gv.y + bv.y;
  y.z = (v[2] - mu) * rstd * gv.z + bv.z;
  y.w = (v[3] - mu) * rstd * gv.w + bv.w;
  ((float4*)(yf + base))[tid] = y;
  if (yb) {
    union { __bf16 b[4]; uint2 u; } pk;
    pk.b[0] = f2bf(y.x); pk.b[1] = f2bf(y.y); pk.b[2] = f2bf(y.z); pk.b[3] = f2bf(y.w);
    ((uint2*)(yb + base))[tid] = pk.u;
  }
}

// ---------------------------------------------------------------------------
// Host launcher
// ---------------------------------------------------------------------------
extern "C" void kernel_launch(void* const* d_in, const int* in_sizes, int n_in,
                              void* d_out, int out_size, void* d_ws, size_t ws_size,
                              hipStream_t stream) {
  const float* tgt      = (const float*)d_in[0];
  const float* enc      = (const float*)d_in[1];
  const int*   src_mask = (const int*)d_in[2];
  const int*   tgt_mask = (const int*)d_in[3];
  const float* sa_wq = (const float*)d_in[4],  *sa_bq = (const float*)d_in[5];
  const float* sa_wk = (const float*)d_in[6],  *sa_bk = (const float*)d_in[7];
  const float* sa_wv = (const float*)d_in[8],  *sa_bv = (const float*)d_in[9];
  const float* sa_wo = (const float*)d_in[10], *sa_bo = (const float*)d_in[11];
  const float* ca_wq = (const float*)d_in[12], *ca_bq = (const float*)d_in[13];
  const float* ca_wk = (const float*)d_in[14], *ca_bk = (const float*)d_in[15];
  const float* ca_wv = (const float*)d_in[16], *ca_bv = (const float*)d_in[17];
  const float* ca_wo = (const float*)d_in[18], *ca_bo = (const float*)d_in[19];
  const float* ffn_w1 = (const float*)d_in[20], *ffn_b1 = (const float*)d_in[21];
  const float* ffn_w2 = (const float*)d_in[22], *ffn_b2 = (const float*)d_in[23];
  const float* ln1_g = (const float*)d_in[24], *ln1_b = (const float*)d_in[25];
  const float* ln2_g = (const float*)d_in[26], *ln2_b = (const float*)d_in[27];
  const float* ln3_g = (const float*)d_in[28], *ln3_b = (const float*)d_in[29];

  // ---- workspace layout (deterministic) ----
  size_t off = 0;
  auto take = [&](size_t bytes) -> void* {
    void* r = (char*)d_ws + off;
    off += (bytes + 255) & ~(size_t)255;
    return r;
  };
  int* flags = (int*)take(256);   // flags[0]=tgt_mask all-ones, flags[1]=src_mask
  const size_t WDD = (size_t)D_MODEL * D_MODEL * 2;  // bf16 bytes
  __bf16* b_sa_wq = (__bf16*)take(WDD);
  __bf16* b_sa_wk = (__bf16*)take(WDD);
  __bf16* b_sa_wv = (__bf16*)take(WDD);
  __bf16* b_sa_wo = (__bf16*)take(WDD);
  __bf16* b_ca_wq = (__bf16*)take(WDD);
  __bf16* b_ca_wk = (__bf16*)take(WDD);
  __bf16* b_ca_wv = (__bf16*)take(WDD);
  __bf16* b_ca_wo = (__bf16*)take(WDD);
  __bf16* b_w1 = (__bf16*)take((size_t)D_FF * D_MODEL * 2);
  __bf16* b_w2 = (__bf16*)take((size_t)D_MODEL * D_FF * 2);

  const size_t ACT2 = (size_t)MROWS * D_MODEL * 2;  // bf16 activation
  const size_t ACT4 = (size_t)MROWS * D_MODEL * 4;  // f32 activation
  __bf16* tgtb = (__bf16*)take(ACT2);
  __bf16* encb = (__bf16*)take(ACT2);
  __bf16* x1b  = (__bf16*)take(ACT2);
  __bf16* x2b  = (__bf16*)take(ACT2);
  __bf16* Qb   = (__bf16*)take(ACT2);
  __bf16* Kb   = (__bf16*)take(ACT2);
  __bf16* Vtb  = (__bf16*)take(ACT2);   // per-head transposed V
  __bf16* Ob   = (__bf16*)take(ACT2);
  float*  t2   = (float*)take(ACT4);
  float*  x1f  = (float*)take(ACT4);
  float*  x2f  = (float*)take(ACT4);
  __bf16* hb   = (__bf16*)take((size_t)MROWS * D_FF * 2);
  (void)ws_size; (void)in_sizes; (void)n_in; (void)out_size;

  auto cvt = [&](const float* in, __bf16* outp, size_t n) {
    cvt_f32_bf16_kernel<<<dim3(1024), dim3(256), 0, stream>>>(in, outp, n / 4);
  };
  auto gemmF = [&](const __bf16* A, const __bf16* W, const float* bias, float* C,
                   int M, int N, int K) {
    gemm_nt_kernel<0, 0><<<dim3(N / 128, M / 128), dim3(128), 0, stream>>>(
        A, W, bias, C, nullptr, M, N, K, 1.0f);
  };
  auto gemmB = [&](const __bf16* A, const __bf16* W, const float* bias, __bf16* C,
                   int M, int N, int K, bool relu, float scale) {
    if (relu)
      gemm_nt_kernel<1, 1><<<dim3(N / 128, M / 128), dim3(128), 0, stream>>>(
          A, W, bias, nullptr, C, M, N, K, scale);
    else
      gemm_nt_kernel<0, 1><<<dim3(N / 128, M / 128), dim3(128), 0, stream>>>(
          A, W, bias, nullptr, C, M, N, K, scale);
  };
  auto gemmV = [&](const __bf16* A, const __bf16* W, const float* bias, __bf16* C) {
    gemm_nt_kernel<0, 2><<<dim3(D_MODEL / 128, MROWS / 128), dim3(128), 0, stream>>>(
        A, W, bias, nullptr, C, MROWS, D_MODEL, D_MODEL, 1.0f);
  };

  // ---- mask pre-scan + precision casts ----
  flag_set_kernel<<<dim3(1), dim3(32), 0, stream>>>(flags);
  mask_check_kernel<<<dim3(1024), dim3(256), 0, stream>>>(
      tgt_mask, (size_t)BATCH * SEQ * SEQ, flags + 0);
  mask_check_kernel<<<dim3(8), dim3(256), 0, stream>>>(
      src_mask, (size_t)BATCH * SEQ, flags + 1);

  cvt(tgt, tgtb, (size_t)MROWS * D_MODEL);
  cvt(enc, encb, (size_t)MROWS * D_MODEL);
  cvt(sa_wq, b_sa_wq, (size_t)D_MODEL * D_MODEL);
  cvt(sa_wk, b_sa_wk, (size_t)D_MODEL * D_MODEL);
  cvt(sa_wv, b_sa_wv, (size_t)D_MODEL * D_MODEL);
  cvt(sa_wo, b_sa_wo, (size_t)D_MODEL * D_MODEL);
  cvt(ca_wq, b_ca_wq, (size_t)D_MODEL * D_MODEL);
  cvt(ca_wk, b_ca_wk, (size_t)D_MODEL * D_MODEL);
  cvt(ca_wv, b_ca_wv, (size_t)D_MODEL * D_MODEL);
  cvt(ca_wo, b_ca_wo, (size_t)D_MODEL * D_MODEL);
  cvt(ffn_w1, b_w1, (size_t)D_FF * D_MODEL);
  cvt(ffn_w2, b_w2, (size_t)D_MODEL * D_FF);

  const dim3 attnGrid(SEQ / 32, BATCH * NUM_HEADS);

  // ---- self attention ----
  gemmB(tgtb, b_sa_wq, sa_bq, Qb, MROWS, D_MODEL, D_MODEL, false, QSCALE);
  gemmB(tgtb, b_sa_wk, sa_bk, Kb, MROWS, D_MODEL, D_MODEL, false, 1.0f);
  gemmV(tgtb, b_sa_wv, sa_bv, Vtb);
  flash_attn_kernel<<<attnGrid, dim3(32), 0, stream>>>(Qb, Kb, Vtb, tgt_mask, 1,
                                                       flags + 0, Ob, SEQ, SEQ);
  gemmF(Ob, b_sa_wo, sa_bo, t2, MROWS, D_MODEL, D_MODEL);
  residual_ln_kernel<<<dim3(MROWS), dim3(256), 0, stream>>>(tgt, t2, ln1_g, ln1_b,
                                                            x1f, x1b);

  // ---- cross attention ----
  gemmB(x1b,  b_ca_wq, ca_bq, Qb, MROWS, D_MODEL, D_MODEL, false, QSCALE);
  gemmB(encb, b_ca_wk, ca_bk, Kb, MROWS, D_MODEL, D_MODEL, false, 1.0f);
  gemmV(encb, b_ca_wv, ca_bv, Vtb);
  flash_attn_kernel<<<attnGrid, dim3(32), 0, stream>>>(Qb, Kb, Vtb, src_mask, 0,
                                                       flags + 1, Ob, SEQ, SEQ);
  gemmF(Ob, b_ca_wo, ca_bo, t2, MROWS, D_MODEL, D_MODEL);
  residual_ln_kernel<<<dim3(MROWS), dim3(256), 0, stream>>>(x1f, t2, ln2_g, ln2_b,
                                                            x2f, x2b);

  // ---- FFN ----
  gemmB(x2b, b_w1, ffn_b1, hb, MROWS, D_FF, D_MODEL, true, 1.0f);   // fused ReLU
  gemmF(hb, b_w2, ffn_b2, t2, MROWS, D_MODEL, D_FF);
  residual_ln_kernel<<<dim3(MROWS), dim3(256), 0, stream>>>(x2f, t2, ln3_g, ln3_b,
                                                            (float*)d_out, nullptr);
}
